// Transformer_vit_30545807409397
// MI455X (gfx1250) — matrix operand order repaired and checked by reading, not compiled, not run on hardware
//
#include <hip/hip_runtime.h>
#include <hip/hip_bf16.h>

typedef unsigned short u16;
typedef __attribute__((ext_vector_type(16))) __bf16 v16bf;
typedef __attribute__((ext_vector_type(8)))  float  v8f;

struct alignas(16) U4x2 { uint4 a, b; };

static __device__ __forceinline__ uint4 ld4(const u16* p) {
  return *reinterpret_cast<const uint4*>(p);
}

static __device__ __forceinline__ v16bf mkfrag(uint4 lo, uint4 hi) {
  U4x2 u; u.a = lo; u.b = hi;
  return __builtin_bit_cast(v16bf, u);
}

static __device__ __forceinline__ u16 f2bf(float f) {  // round-to-nearest-even
  unsigned x = __builtin_bit_cast(unsigned, f);
  unsigned r = x + 0x7FFFu + ((x >> 16) & 1u);
  return (u16)(r >> 16);
}

static __device__ __forceinline__ v8f wmma_bf16(v16bf a, v16bf b, v8f c) {
  return __builtin_amdgcn_wmma_f32_16x16x32_bf16(false, a, false, b, (short)0, c,
                                                 false, false);
}

static __device__ __forceinline__ float wave_sum(float v) {
#pragma unroll
  for (int off = 16; off; off >>= 1) v += __shfl_xor(v, off, 32);
  return v;
}

// ---------------------------------------------------------------------------
// Weight pack: W (f32, K x N row-major) -> Wt (bf16, N x K row-major)
// ---------------------------------------------------------------------------
__global__ void pack_wt(const float* __restrict__ W, u16* __restrict__ Wt,
                        int K, int N) {
  int idx = blockIdx.x * 256 + threadIdx.x;
  if (idx >= K * N) return;
  int n = idx / K, k = idx % K;
  Wt[idx] = f2bf(W[(size_t)k * N + n]);
}

// ---------------------------------------------------------------------------
// tok = transpose(x) + pos (f32), ln1 = LayerNorm(tok)*g+b (bf16)
// one wave per token, lane covers channels lane+32j
// ---------------------------------------------------------------------------
__global__ __launch_bounds__(256) void build_tok_ln1(
    const float* __restrict__ x, const float* __restrict__ pos,
    const float* __restrict__ g, const float* __restrict__ beta,
    float* __restrict__ tok, u16* __restrict__ lno) {
  int wave = threadIdx.x >> 5, lane = threadIdx.x & 31;
  int token = blockIdx.x * 8 + wave;
  int b = token >> 10, n = token & 1023;
  float vals[8]; float s = 0.f, q = 0.f;
#pragma unroll
  for (int j = 0; j < 8; ++j) {
    int c = lane + 32 * j;
    float v = x[((size_t)b * 256 + c) * 1024 + n] + pos[(size_t)n * 256 + c];
    vals[j] = v; s += v; q += v * v;
    tok[(size_t)token * 256 + c] = v;
  }
  s = wave_sum(s); q = wave_sum(q);
  float mu = s * (1.f / 256.f);
  float var = q * (1.f / 256.f) - mu * mu;
  float rs = rsqrtf(var + 1e-5f);
#pragma unroll
  for (int j = 0; j < 8; ++j) {
    int c = lane + 32 * j;
    lno[(size_t)token * 256 + c] = f2bf((vals[j] - mu) * rs * g[c] + beta[c]);
  }
}

// LayerNorm only (tok2 f32 -> bf16)
__global__ __launch_bounds__(256) void ln_only(
    const float* __restrict__ in, const float* __restrict__ g,
    const float* __restrict__ beta, u16* __restrict__ lno) {
  int wave = threadIdx.x >> 5, lane = threadIdx.x & 31;
  int token = blockIdx.x * 8 + wave;
  float vals[8]; float s = 0.f, q = 0.f;
#pragma unroll
  for (int j = 0; j < 8; ++j) {
    int c = lane + 32 * j;
    float v = in[(size_t)token * 256 + c];
    vals[j] = v; s += v; q += v * v;
  }
  s = wave_sum(s); q = wave_sum(q);
  float mu = s * (1.f / 256.f);
  float var = q * (1.f / 256.f) - mu * mu;
  float rs = rsqrtf(var + 1e-5f);
#pragma unroll
  for (int j = 0; j < 8; ++j) {
    int c = lane + 32 * j;
    lno[(size_t)token * 256 + c] = f2bf((vals[j] - mu) * rs * g[c] + beta[c]);
  }
}

// ---------------------------------------------------------------------------
// Vt[b][h][d][key] (bf16) from qkv (bf16, 16384 x 384, V at col 256+h*32+d)
// ---------------------------------------------------------------------------
__global__ void pack_vt(const u16* __restrict__ qkv, u16* __restrict__ Vt) {
  int idx = blockIdx.x * 256 + threadIdx.x;  // 16*4*32*1024
  int key = idx & 1023;
  int rest = idx >> 10;
  int d = rest & 31;
  int bh = rest >> 5;
  int b = bh >> 2, h = bh & 3;
  Vt[idx] = qkv[((size_t)(b * 1024 + key)) * 384 + 256 + h * 32 + d];
}

// ---------------------------------------------------------------------------
// Generic bf16 WMMA GEMM: C[16384 x ldN] = A[16384 x KDIM] * Bt^T
//   A row-major bf16, Bt = B^T (ldN x KDIM) row-major bf16.
// Block = 128 thr (4 waves); wave computes 32 rows x 64 cols (8 WMMA / kstep,
// each B fragment reused by two row tiles).
// EPI: 0 = store bf16; 1 = +bias+resid -> f32;
//      2 = gelu(+bias) -> bf16; 3 = +bias+resid -> f32 transposed (b,c,n)
// ---------------------------------------------------------------------------
template <int KDIM, int EPI>
__global__ __launch_bounds__(128) void gemm_bf16_wmma(
    const u16* __restrict__ A, const u16* __restrict__ Bt,
    const float* __restrict__ bias, const float* __restrict__ resid,
    u16* __restrict__ outB, float* __restrict__ outF, int ldN) {
  const int lane = threadIdx.x & 31;
  const int wave = threadIdx.x >> 5;
  const int hi = lane >> 4;
  const int l15 = lane & 15;
  const int rowBase = blockIdx.y * 128 + wave * 32;
  const int nBase = blockIdx.x * 64;

  v8f acc[2][4];
#pragma unroll
  for (int m = 0; m < 2; ++m)
#pragma unroll
    for (int t = 0; t < 4; ++t) acc[m][t] = (v8f){};

  const u16* arow0 = A + (size_t)(rowBase + l15) * KDIM + hi * 8;
  const u16* arow1 = arow0 + (size_t)16 * KDIM;
  const u16* bp[4];
#pragma unroll
  for (int t = 0; t < 4; ++t)
    bp[t] = Bt + (size_t)(nBase + t * 16 + l15) * KDIM + hi * 8;

#pragma unroll
  for (int k0 = 0; k0 < KDIM; k0 += 32) {
    v16bf a0 = mkfrag(ld4(arow0 + k0), ld4(arow0 + k0 + 16));
    v16bf a1 = mkfrag(ld4(arow1 + k0), ld4(arow1 + k0 + 16));
#pragma unroll
    for (int t = 0; t < 4; ++t) {
      v16bf b = mkfrag(ld4(bp[t] + k0), ld4(bp[t] + k0 + 16));
      acc[0][t] = wmma_bf16(a0, b, acc[0][t]);
      acc[1][t] = wmma_bf16(a1, b, acc[1][t]);
    }
  }

#pragma unroll
  for (int m = 0; m < 2; ++m) {
#pragma unroll
    for (int t = 0; t < 4; ++t) {
      const int col = nBase + t * 16 + l15;
#pragma unroll
      for (int r = 0; r < 8; ++r) {
        const int row = rowBase + m * 16 + r + 8 * hi;
        float v = acc[m][t][r];
        if (EPI == 0) {
          outB[(size_t)row * ldN + col] = f2bf(v);
        } else if (EPI == 1) {
          outF[(size_t)row * ldN + col] =
              v + bias[col] + resid[(size_t)row * ldN + col];
        } else if (EPI == 2) {
          float xx = v + bias[col];
          float gl = 0.5f * xx * (1.0f + erff(xx * 0.70710678118654752f));
          outB[(size_t)row * ldN + col] = f2bf(gl);
        } else {
          int bb = row >> 10, n = row & 1023;
          float o = v + bias[col] + resid[(size_t)row * 256 + col];
          outF[(((size_t)(bb * 256 + col)) << 10) + n] = o;
        }
      }
    }
  }
}

// ---------------------------------------------------------------------------
// Flash attention. Grid (64 bh, 16 qblocks), block 128 (4 waves),
// wave owns 16 query rows; loop over 1024 keys in blocks of 32.
// K/V fragments are software-pipelined one iteration ahead so the global
// loads overlap the softmax VALU work ('& 1023' wraps the last prefetch).
// ---------------------------------------------------------------------------
__global__ __launch_bounds__(128) void attn_flash(
    const u16* __restrict__ qkv, const u16* __restrict__ Vt,
    u16* __restrict__ O) {
  __shared__ float Sld[4][16][33];
  __shared__ alignas(16) u16 Pld[4][16][40];
  __shared__ float Al[4][16];
  __shared__ float Ll[4][16];

  const int lane = threadIdx.x & 31, wave = threadIdx.x >> 5;
  const int hi = lane >> 4, l15 = lane & 15;
  const int bh = blockIdx.x, b = bh >> 2, h = bh & 3;
  const int qtok = b * 1024 + blockIdx.y * 64 + wave * 16;

  // Q A-fragment (constant over the key loop)
  const u16* qrow = qkv + (size_t)(qtok + l15) * 384 + h * 32 + hi * 8;
  v16bf aq = mkfrag(ld4(qrow), ld4(qrow + 16));

  v8f o0 = (v8f){}, o1 = (v8f){};
  float mst = -3.0e38f, lst = 0.f;
  const float scale = 0.17677669529663687f;  // 1/sqrt(32)

  const u16* kbase = qkv + (size_t)(b * 1024) * 384 + 128 + h * 32 + hi * 8;
  const u16* vt0 = Vt + ((size_t)bh * 32 + l15) * 1024 + hi * 8;
  const u16* vt1 = vt0 + (size_t)16 * 1024;

  // prefetch fragments for kb = 0
  const u16* kr = kbase + (size_t)l15 * 384;
  uint4 k0a = ld4(kr),            k0b = ld4(kr + 16);
  uint4 k1a = ld4(kr + 16 * 384), k1b = ld4(kr + 16 * 384 + 16);
  uint4 v0a = ld4(vt0),           v0b = ld4(vt0 + 16);
  uint4 v1a = ld4(vt1),           v1b = ld4(vt1 + 16);

  for (int kb = 0; kb < 1024; kb += 32) {
    v16bf bk0 = mkfrag(k0a, k0b);
    v16bf bk1 = mkfrag(k1a, k1b);
    v16bf bv0 = mkfrag(v0a, v0b);
    v16bf bv1 = mkfrag(v1a, v1b);

    // issue next iteration's loads now (wraps to block 0 on last iteration)
    {
      int kn = (kb + 32) & 1023;
      const u16* krn = kbase + (size_t)(kn + l15) * 384;
      k0a = ld4(krn);            k0b = ld4(krn + 16);
      k1a = ld4(krn + 16 * 384); k1b = ld4(krn + 16 * 384 + 16);
      v0a = ld4(vt0 + kn);       v0b = ld4(vt0 + kn + 16);
      v1a = ld4(vt1 + kn);       v1b = ld4(vt1 + kn + 16);
    }

    v8f z = (v8f){};
    v8f s0 = wmma_bf16(aq, bk0, z);
    v8f s1 = wmma_bf16(aq, bk1, z);
#pragma unroll
    for (int r = 0; r < 8; ++r) {
      int row = r + 8 * hi;
      Sld[wave][row][l15] = s0[r] * scale;
      Sld[wave][row][16 + l15] = s1[r] * scale;
    }
    asm volatile("s_wait_dscnt 0" ::: "memory");
    if (lane < 16) {  // lane owns query row `lane`
      float rm = -3.0e38f;
#pragma unroll
      for (int j = 0; j < 32; ++j) rm = fmaxf(rm, Sld[wave][lane][j]);
      float mn = fmaxf(mst, rm);
      float alpha = __expf(mst - mn);
      float ps = 0.f;
#pragma unroll
      for (int j = 0; j < 32; ++j) {
        float p = __expf(Sld[wave][lane][j] - mn);
        ps += p;
        Pld[wave][lane][j] = f2bf(p);
      }
      lst = lst * alpha + ps;
      mst = mn;
      Al[wave][lane] = alpha;
    }
    asm volatile("s_wait_dscnt 0" ::: "memory");
#pragma unroll
    for (int r = 0; r < 8; ++r) {
      float a = Al[wave][r + 8 * hi];
      o0[r] *= a; o1[r] *= a;
    }
    const u16* pr = &Pld[wave][l15][hi * 8];
    v16bf pf = mkfrag(ld4(pr), ld4(pr + 16));
    o0 = wmma_bf16(pf, bv0, o0);
    o1 = wmma_bf16(pf, bv1, o1);
  }
  if (lane < 16) Ll[wave][lane] = lst;
  asm volatile("s_wait_dscnt 0" ::: "memory");
#pragma unroll
  for (int r = 0; r < 8; ++r) {
    float inv = 1.0f / Ll[wave][r + 8 * hi];
    int tokg = qtok + r + 8 * hi;
    O[(size_t)tokg * 128 + h * 32 + l15] = f2bf(o0[r] * inv);
    O[(size_t)tokg * 128 + h * 32 + 16 + l15] = f2bf(o1[r] * inv);
  }
}

// ---------------------------------------------------------------------------
extern "C" void kernel_launch(void* const* d_in, const int* in_sizes, int n_in,
                              void* d_out, int out_size, void* d_ws,
                              size_t ws_size, hipStream_t stream) {
  (void)in_sizes; (void)n_in; (void)out_size; (void)ws_size;
  const float* x     = (const float*)d_in[0];
  const float* pos   = (const float*)d_in[1];
  const float* w_qkv = (const float*)d_in[2];
  const float* w_out = (const float*)d_in[3];
  const float* b_out = (const float*)d_in[4];
  const float* w_ff1 = (const float*)d_in[5];
  const float* b_ff1 = (const float*)d_in[6];
  const float* w_ff2 = (const float*)d_in[7];
  const float* b_ff2 = (const float*)d_in[8];
  const float* g1    = (const float*)d_in[9];
  const float* bt1   = (const float*)d_in[10];
  const float* g2    = (const float*)d_in[11];
  const float* bt2   = (const float*)d_in[12];

  char* w = (char*)d_ws;
  float* tok  = (float*)w; w += (size_t)16384 * 256 * 4;   // 16 MB
  float* tok2 = (float*)w; w += (size_t)16384 * 256 * 4;   // 16 MB
  u16* ln1    = (u16*)w;   w += (size_t)16384 * 256 * 2;   // 8 MB
  u16* ln2b   = (u16*)w;   w += (size_t)16384 * 256 * 2;   // 8 MB
  u16* qkvB   = (u16*)w;   w += (size_t)16384 * 384 * 2;   // 12 MB
  u16* Vt     = (u16*)w;   w += (size_t)64 * 32 * 1024 * 2; // 4 MB
  u16* ao     = (u16*)w;   w += (size_t)16384 * 128 * 2;   // 4 MB
  u16* hdn    = (u16*)w;   w += (size_t)16384 * 128 * 2;   // 4 MB
  u16* WqkvT  = (u16*)w;   w += (size_t)384 * 256 * 2;
  u16* WoutT  = (u16*)w;   w += (size_t)256 * 128 * 2;
  u16* Wff1T  = (u16*)w;   w += (size_t)128 * 256 * 2;
  u16* Wff2T  = (u16*)w;   w += (size_t)256 * 128 * 2;

  // 1. weight packs (bf16, transposed)
  pack_wt<<<(256 * 384 + 255) / 256, 256, 0, stream>>>(w_qkv, WqkvT, 256, 384);
  pack_wt<<<(128 * 256 + 255) / 256, 256, 0, stream>>>(w_out, WoutT, 128, 256);
  pack_wt<<<(256 * 128 + 255) / 256, 256, 0, stream>>>(w_ff1, Wff1T, 256, 128);
  pack_wt<<<(128 * 256 + 255) / 256, 256, 0, stream>>>(w_ff2, Wff2T, 128, 256);

  // 2. tok = x^T + pos ; ln1
  build_tok_ln1<<<2048, 256, 0, stream>>>(x, pos, g1, bt1, tok, ln1);

  // 3. qkv = ln1 @ w_qkv  (M=16384, K=256, N=384) -> bf16
  gemm_bf16_wmma<256, 0><<<dim3(6, 128), 128, 0, stream>>>(
      ln1, WqkvT, nullptr, nullptr, qkvB, nullptr, 384);

  // 4. V transpose pack
  pack_vt<<<8192, 256, 0, stream>>>(qkvB, Vt);

  // 5. flash attention -> ao (bf16, 16384 x 128)
  attn_flash<<<dim3(64, 16), 128, 0, stream>>>(qkvB, Vt, ao);

  // 6. tok2 = ao @ w_out + b_out + tok  (K=128, N=256) -> f32
  gemm_bf16_wmma<128, 1><<<dim3(4, 128), 128, 0, stream>>>(
      ao, WoutT, b_out, tok, nullptr, tok2, 256);

  // 7. ln2
  ln_only<<<2048, 256, 0, stream>>>(tok2, g2, bt2, ln2b);

  // 8. hdn = gelu(ln2 @ w_ff1 + b_ff1)  (K=256, N=128) -> bf16
  gemm_bf16_wmma<256, 2><<<dim3(2, 128), 128, 0, stream>>>(
      ln2b, Wff1T, b_ff1, nullptr, hdn, nullptr, 128);

  // 9. out = hdn @ w_ff2 + b_ff2 + tok2, transposed to (b,c,h,w)
  gemm_bf16_wmma<128, 3><<<dim3(4, 128), 128, 0, stream>>>(
      hdn, Wff2T, b_ff2, tok2, nullptr, (float*)d_out, 256);
}
